// RowWiseGateSelfAttention_21088289423631
// MI455X (gfx1250) — matrix operand order repaired and checked by reading, not compiled, not run on hardware
//
#include <hip/hip_runtime.h>
#include <math.h>

// ---------------- constants ----------------
#define BB 2
#define LL 512
#define CC 576
#define HH 8
#define CPP 128
#define C3 192
#define DK3 24
#define DKP 32
#define DIVI 0.117851130197757921f   // 1/sqrt(72)

// workspace byte offsets
#define NQKV ((size_t)BB*3*HH*LL*DKP)          // 786432 bf16 elems
#define OFF_Q   ((size_t)0)
#define OFF_K   (OFF_Q + NQKV*2)
#define OFF_V   (OFF_K + NQKV*2)
#define OFF_G   (OFF_V + NQKV*2)               // gate: B*3*H*L*24 f32
#define OFF_BIAS (OFF_G + (size_t)BB*3*HH*LL*DK3*4)
#define OFF_ORI  (OFF_BIAS + (size_t)BB*HH*LL*LL*4)
#define OFF_CONV (OFF_ORI + (size_t)BB*3*HH*LL*LL*4)
#define OFF_HO   (OFF_CONV + (size_t)BB*HH*LL*LL*4)

typedef __attribute__((ext_vector_type(16))) __bf16 v16bf;
typedef __attribute__((ext_vector_type(8)))  float  v8f;

union ABf { unsigned int u[8]; v16bf v; };

__device__ __forceinline__ unsigned short f2bf(float f) {
  union { float f; unsigned u; } a; a.f = f;
  unsigned r = a.u + 0x7FFFu + ((a.u >> 16) & 1u);
  return (unsigned short)(r >> 16);
}

__device__ __forceinline__ v8f wmma_bf16(const ABf& a, const ABf& b, v8f c) {
  return __builtin_amdgcn_wmma_f32_16x16x32_bf16(false, a.v, false, b.v,
                                                 (short)0, c, false, false);
}

// ---------------- kernel 1: LN(x) + q/k/v/g projections ----------------
__global__ __launch_bounds__(192) void k_ln_proj(
    const float* __restrict__ x, const float* __restrict__ ln1w, const float* __restrict__ ln1b,
    const float* __restrict__ Wq, const float* __restrict__ bq,
    const float* __restrict__ Wk, const float* __restrict__ bk,
    const float* __restrict__ Wv, const float* __restrict__ bv,
    const float* __restrict__ Wg, const float* __restrict__ bg,
    unsigned short* __restrict__ qo, unsigned short* __restrict__ ko,
    unsigned short* __restrict__ vo, float* __restrict__ go)
{
  int row = blockIdx.x;              // b*L + l
  int t = threadIdx.x;               // 0..191
  __shared__ float xs[CC];
  __shared__ float red[8];
  const float* xr = x + (size_t)row * CC;

  float s0 = 0.f;
  for (int i = t; i < CC; i += 192) { float v = xr[i]; xs[i] = v; s0 += v; }
  for (int m = 16; m > 0; m >>= 1) s0 += __shfl_xor(s0, m, 32);
  if ((t & 31) == 0) red[t >> 5] = s0;
  __syncthreads();
  float mean = 0.f;
  for (int w = 0; w < 6; w++) mean += red[w];
  mean *= (1.f / CC);
  __syncthreads();

  float s1 = 0.f;
  for (int i = t; i < CC; i += 192) { float d = xs[i] - mean; s1 += d * d; }
  for (int m = 16; m > 0; m >>= 1) s1 += __shfl_xor(s1, m, 32);
  if ((t & 31) == 0) red[t >> 5] = s1;
  __syncthreads();
  float var = 0.f;
  for (int w = 0; w < 6; w++) var += red[w];
  var *= (1.f / CC);
  float rs = rsqrtf(var + 1e-6f);
  __syncthreads();

  for (int i = t; i < CC; i += 192) xs[i] = (xs[i] - mean) * rs * ln1w[i] + ln1b[i];
  __syncthreads();

  int o = t;                    // output channel within group 0..191
  int h = o / DK3, d = o % DK3;
  int b = row >> 9, l = row & 511;
  for (int c = 0; c < 3; c++) {
    const float* xg = xs + c * C3;
    float aq = 0.f, ak = 0.f, av = 0.f, ag = 0.f;
    for (int i = 0; i < C3; i++) {
      float xv = xg[i];
      aq += xv * Wq[i * C3 + o];
      ak += xv * Wk[i * C3 + o];
      av += xv * Wv[i * C3 + o];
      ag += xv * Wg[i * C3 + o];
    }
    aq += bq[o]; ak += bk[o]; av += bv[o]; ag += bg[o];
    int bch = (b * 3 + c) * HH + h;
    size_t idx = ((size_t)bch * LL + l) * DKP + d;
    qo[idx] = f2bf(aq); ko[idx] = f2bf(ak); vo[idx] = f2bf(av);
    if (d < 8) { qo[idx + 24] = 0; ko[idx + 24] = 0; vo[idx + 24] = 0; }
    size_t gidx = ((size_t)bch * LL + l) * DK3 + d;
    go[gidx] = 1.f / (1.f + expf(-ag));
  }
}

// ---------------- kernel 2: LN(pairX) + pair bias (one wave per position) ---
__global__ __launch_bounds__(256) void k_pair_bias(
    const float* __restrict__ pairX, const float* __restrict__ ln2w,
    const float* __restrict__ ln2b, const float* __restrict__ Wp,
    const float* __restrict__ bp, float* __restrict__ bias)
{
  int wv = threadIdx.x >> 5;
  int lane = threadIdx.x & 31;
  int wid = blockIdx.x * 8 + wv;           // == (b*L + i)*L + j
  const float* pr = pairX + (size_t)wid * CPP;
  __builtin_prefetch(pr + 8 * CPP, 0, 1);

  float4 v = *(const float4*)(pr + lane * 4);
  float s = v.x + v.y + v.z + v.w;
  float s2 = v.x * v.x + v.y * v.y + v.z * v.z + v.w * v.w;
  for (int m = 16; m > 0; m >>= 1) { s += __shfl_xor(s, m, 32); s2 += __shfl_xor(s2, m, 32); }
  float mean = s * (1.f / CPP);
  float var = s2 * (1.f / CPP) - mean * mean;
  float rs = rsqrtf(var + 1e-6f);

  int c0 = lane * 4;
  float p0 = (v.x - mean) * rs * ln2w[c0 + 0] + ln2b[c0 + 0];
  float p1 = (v.y - mean) * rs * ln2w[c0 + 1] + ln2b[c0 + 1];
  float p2 = (v.z - mean) * rs * ln2w[c0 + 2] + ln2b[c0 + 2];
  float p3 = (v.w - mean) * rs * ln2w[c0 + 3] + ln2b[c0 + 3];

  float acc[HH];
  #pragma unroll
  for (int h = 0; h < HH; h++) {
    acc[h] = p0 * Wp[(c0 + 0) * HH + h] + p1 * Wp[(c0 + 1) * HH + h]
           + p2 * Wp[(c0 + 2) * HH + h] + p3 * Wp[(c0 + 3) * HH + h];
    for (int m = 16; m > 0; m >>= 1) acc[h] += __shfl_xor(acc[h], m, 32);
  }

  __shared__ float mb[8][HH];
  if (lane < HH) mb[wv][lane] = acc[lane] + bp[lane];
  __syncthreads();

  int t = threadIdx.x;
  if (t < 64) {
    int h = t >> 3, w = t & 7;
    int wj = blockIdx.x * 8 + w;
    int j = wj & 511, i = (wj >> 9) & 511, b = wj >> 18;
    bias[(((size_t)b * HH + h) * LL + i) * LL + j] = mb[w][h];
  }
}

// ---------------- kernel 3: S = Q K^T + bias (WMMA, one wave per tile) -----
__global__ __launch_bounds__(128) void k_qk(
    const unsigned short* __restrict__ q, const unsigned short* __restrict__ k,
    const float* __restrict__ bias, float* __restrict__ ori)
{
  int wid = blockIdx.x * 4 + (threadIdx.x >> 5);
  int lane = threadIdx.x & 31;
  int bch = wid >> 10;
  int tt = wid & 1023;
  int ti = tt >> 5, tj = tt & 31;
  int b = bch / 24, h = bch & 7;
  size_t base = (size_t)bch * (LL * DKP);
  const unsigned short* qp = q + base;
  const unsigned short* kp = k + base;
  int i0 = ti * 16, j0 = tj * 16;
  int half = lane >> 4, m = lane & 15;

  ABf A, Bm;
  #pragma unroll
  for (int jv = 0; jv < 8; jv++) {           // A: 16x32 bf16 layout
    int k0 = (jv < 4 ? 2 * jv : 16 + 2 * (jv - 4)) + 8 * half;
    A.u[jv] = *(const unsigned int*)(qp + (size_t)(i0 + m) * DKP + k0);
  }
  #pragma unroll
  for (int jv = 0; jv < 8; jv++) {           // B: 32x16 bf16 layout, same K order as A
    int k0 = (jv < 4 ? 2 * jv : 16 + 2 * (jv - 4)) + 8 * half;
    Bm.u[jv] = *(const unsigned int*)(kp + (size_t)(j0 + m) * DKP + k0);
  }
  v8f acc = {};
  acc = wmma_bf16(A, Bm, acc);

  const float* bptr = bias + ((size_t)b * HH + h) * LL * LL;
  float* optr = ori + (size_t)bch * LL * LL;
  int n = lane & 15, mb = half * 8;
  #pragma unroll
  for (int r = 0; r < 8; r++) {
    int mi = i0 + mb + r, nj = j0 + n;
    optr[(size_t)mi * LL + nj] = acc[r] + bptr[(size_t)mi * LL + nj];
  }
}

// ---------------- kernel 4: 5x5 conv refinement, summed over 3 branches ----
__global__ __launch_bounds__(256) void k_conv(
    const float* __restrict__ ori, const float* __restrict__ cw,
    const float* __restrict__ cb, float* __restrict__ convq)
{
  int blk = blockIdx.x;
  int bh = blk >> 10, tt = blk & 1023;
  int ti = tt >> 5, tj = tt & 31;
  int b = bh >> 3, ho = bh & 7;
  int i0 = ti * 16, j0 = tj * 16;
  int tx = threadIdx.x & 15, ty = threadIdx.x >> 4;
  __shared__ float tile[20][20];
  float acc = 0.f;

  for (int c = 0; c < 3; c++)
    for (int hi = 0; hi < HH; hi++) {
      __syncthreads();
      for (int idx = threadIdx.x; idx < 400; idx += 256) {
        int r = idx / 20, col = idx % 20;
        int gi = i0 - 2 + r, gj = j0 - 2 + col;
        float v = 0.f;
        if (gi >= 0 && gi < LL && gj >= 0 && gj < LL)
          v = ori[(((size_t)(b * 3 + c) * HH + hi) * LL + gi) * LL + gj];
        tile[r][col] = v;
      }
      __syncthreads();
      const float* w = cw + ((c * HH + ho) * HH + hi) * 25;
      #pragma unroll
      for (int dy = 0; dy < 5; dy++)
        #pragma unroll
        for (int dx = 0; dx < 5; dx++)
          acc += w[dy * 5 + dx] * tile[ty + dy][tx + dx];
    }
  acc += cb[ho] + cb[HH + ho] + cb[2 * HH + ho];
  convq[(((size_t)b * HH + ho) * LL + i0 + ty) * LL + j0 + tx] = acc;
}

// ---------------- kernel 5: softmax (LDS-staged bf16) + score@V (WMMA) -----
// 2 waves / block, 32KB LDS prob stage. Pass 1 reads logits ONCE, stages
// exp(x-max) as bf16 rows in LDS; pass 2 builds WMMA A-fragments from LDS and
// folds 1/sum + sigmoid(gate) into the D-matrix epilogue.
__global__ __launch_bounds__(64) void k_attn(
    const float* __restrict__ ori, const float* __restrict__ convq,
    const unsigned short* __restrict__ vmat, const float* __restrict__ gate,
    float* __restrict__ hout)
{
  int wv = threadIdx.x >> 5;
  int lane = threadIdx.x & 31;
  int wid = blockIdx.x * 2 + wv;
  int bch = wid >> 5, ti = wid & 31;
  int b = bch / 24, c = (bch / 8) % 3, h = bch & 7;
  const float* op = ori + (size_t)bch * LL * LL;
  const float* cp = convq + ((size_t)b * HH + h) * LL * LL;
  int i0 = ti * 16;
  int half = lane >> 4;

  __shared__ unsigned short probs[2][16][LL];   // 32 KB

  float rinv_arr[8];                            // 1/sum for rows half*8+r

  for (int m = 0; m < 16; m++) {                // ---- pass 1 ----
    size_t roff = (size_t)(i0 + m) * LL + lane * 16;
    float vals[16];
    #pragma unroll
    for (int u = 0; u < 4; u++) {
      float4 a = *(const float4*)(op + roff + 4 * u);
      float4 d = *(const float4*)(cp + roff + 4 * u);
      vals[4 * u + 0] = (a.x + d.x) * DIVI;
      vals[4 * u + 1] = (a.y + d.y) * DIVI;
      vals[4 * u + 2] = (a.z + d.z) * DIVI;
      vals[4 * u + 3] = (a.w + d.w) * DIVI;
    }
    float mx = vals[0];
    #pragma unroll
    for (int t2 = 1; t2 < 16; t2++) mx = fmaxf(mx, vals[t2]);
    for (int s = 16; s > 0; s >>= 1) mx = fmaxf(mx, __shfl_xor(mx, s, 32));
    float sm = 0.f;
    unsigned pk[8];
    #pragma unroll
    for (int t2 = 0; t2 < 8; t2++) {
      float e0 = expf(vals[2 * t2] - mx);
      float e1 = expf(vals[2 * t2 + 1] - mx);
      sm += e0 + e1;
      pk[t2] = (unsigned)f2bf(e0) | ((unsigned)f2bf(e1) << 16);
    }
    for (int s = 16; s > 0; s >>= 1) sm += __shfl_xor(sm, s, 32);
    if ((m >> 3) == half) rinv_arr[m & 7] = 1.f / sm;
    uint4* dst = (uint4*)&probs[wv][m][lane * 16];
    dst[0] = make_uint4(pk[0], pk[1], pk[2], pk[3]);
    dst[1] = make_uint4(pk[4], pk[5], pk[6], pk[7]);
  }
  __syncthreads();

  int mrow = lane & 15;
  size_t vbase = (size_t)bch * LL * DKP;
  v8f acc0 = {}, acc1 = {};

  for (int s = 0; s < 16; s++) {                // ---- pass 2: 32-wide K ----
    int jc = s * 32;
    ABf A, B0, B1;
    const unsigned short* prow = &probs[wv][mrow][0];
    #pragma unroll
    for (int jv = 0; jv < 8; jv++) {
      int k0 = (jv < 4 ? 2 * jv : 16 + 2 * (jv - 4)) + 8 * half;
      A.u[jv] = *(const unsigned int*)(prow + jc + k0);
    }
    const unsigned short* vp = vmat + vbase;
    #pragma unroll
    for (int jv = 0; jv < 8; jv++) {
      int k0 = (jv < 4 ? 2 * jv : 16 + 2 * (jv - 4)) + 8 * half;
      size_t r0 = (size_t)(jc + k0) * DKP, r1 = (size_t)(jc + k0 + 1) * DKP;
      B0.u[jv] = (unsigned)vp[r0 + mrow] | ((unsigned)vp[r1 + mrow] << 16);
      B1.u[jv] = (unsigned)vp[r0 + 16 + mrow] | ((unsigned)vp[r1 + 16 + mrow] << 16);
    }
    acc0 = wmma_bf16(A, B0, acc0);
    acc1 = wmma_bf16(A, B1, acc1);
  }

  int n = lane & 15, mb = half * 8;
  const float* gp = gate + (size_t)bch * LL * DK3;
  int cbase = c * C3 + h * DK3;
  #pragma unroll
  for (int r = 0; r < 8; r++) {
    int mi = i0 + mb + r;
    float rinv = rinv_arr[r];
    float o0 = acc0[r] * rinv * gp[(size_t)mi * DK3 + n];
    hout[((size_t)b * LL + mi) * CC + cbase + n] = o0;
    if (n < 8) {
      float o1 = acc1[r] * rinv * gp[(size_t)mi * DK3 + 16 + n];
      hout[((size_t)b * LL + mi) * CC + cbase + 16 + n] = o1;
    }
  }
}

// ---------------- kernel 6: gelu(mean qk) @ Wpr -> out2 --------------------
__global__ __launch_bounds__(256) void k_out2(
    const float* __restrict__ ori, const float* __restrict__ convq,
    const float* __restrict__ Wpr, const float* __restrict__ bpr,
    float* __restrict__ out2)
{
  int blk = blockIdx.x;
  int j0 = (blk & 63) * 8, i = (blk >> 6) & 511, b = blk >> 15;
  __shared__ float mg[64];
  __shared__ float wpr_s[HH * CPP];
  __shared__ float bpr_s[CPP];
  int t = threadIdx.x;
  for (int idx = t; idx < HH * CPP; idx += 256) wpr_s[idx] = Wpr[idx];
  if (t < CPP) bpr_s[t] = bpr[t];
  if (t < 64) {
    int jj = t >> 3, h = t & 7;
    size_t pos = (size_t)i * LL + (j0 + jj);
    float s = 0.f;
    for (int c = 0; c < 3; c++) s += ori[((size_t)(b * 3 + c) * HH + h) * LL * LL + pos];
    float cv = convq[((size_t)b * HH + h) * LL * LL + pos];
    float x = (s * (1.f / 3.f) + cv) * DIVI;
    mg[jj * 8 + h] = 0.5f * x * (1.f + erff(x * 0.70710678118654752f));
  }
  __syncthreads();

  int jj = t >> 5, lane = t & 31;
  int c0 = lane * 4;
  float m0 = mg[jj * 8 + 0], m1 = mg[jj * 8 + 1], m2 = mg[jj * 8 + 2], m3 = mg[jj * 8 + 3];
  float m4 = mg[jj * 8 + 4], m5 = mg[jj * 8 + 5], m6 = mg[jj * 8 + 6], m7 = mg[jj * 8 + 7];
  float4 r;
  float* rr = (float*)&r;
  #pragma unroll
  for (int u = 0; u < 4; u++) {
    int cc = c0 + u;
    rr[u] = bpr_s[cc]
          + m0 * wpr_s[0 * CPP + cc] + m1 * wpr_s[1 * CPP + cc]
          + m2 * wpr_s[2 * CPP + cc] + m3 * wpr_s[3 * CPP + cc]
          + m4 * wpr_s[4 * CPP + cc] + m5 * wpr_s[5 * CPP + cc]
          + m6 * wpr_s[6 * CPP + cc] + m7 * wpr_s[7 * CPP + cc];
  }
  float* o = out2 + (((size_t)b * LL + i) * LL + (j0 + jj)) * CPP + c0;
  *(float4*)o = r;
}

// ---------------- kernel 7: out1 = h_outs @ Wo + bo + x*g ------------------
__global__ __launch_bounds__(192) void k_out1(
    const float* __restrict__ hout, const float* __restrict__ Wo,
    const float* __restrict__ bo, const float* __restrict__ x,
    const float* __restrict__ g, float* __restrict__ out1)
{
  int row = blockIdx.x, t = threadIdx.x;
  __shared__ float hs[CC];
  const float* hr = hout + (size_t)row * CC;
  for (int i = t; i < CC; i += 192) hs[i] = hr[i];
  __syncthreads();
  float gv = g[0];
  #pragma unroll
  for (int oo = 0; oo < 3; oo++) {
    int o = t + oo * 192;
    float acc = 0.f;
    for (int i = 0; i < CC; i++) acc += hs[i] * Wo[(size_t)i * CC + o];
    out1[(size_t)row * CC + o] = acc + bo[o] + x[(size_t)row * CC + o] * gv;
  }
}

// ---------------- launcher ----------------
extern "C" void kernel_launch(void* const* d_in, const int* in_sizes, int n_in,
                              void* d_out, int out_size, void* d_ws, size_t ws_size,
                              hipStream_t stream) {
  (void)in_sizes; (void)n_in; (void)out_size; (void)ws_size;
  const float* x     = (const float*)d_in[0];
  const float* pairX = (const float*)d_in[1];
  const float* g     = (const float*)d_in[2];
  const float* ln1w  = (const float*)d_in[3];
  const float* ln1b  = (const float*)d_in[4];
  const float* ln2w  = (const float*)d_in[5];
  const float* ln2b  = (const float*)d_in[6];
  const float* Wq = (const float*)d_in[7],  *bq = (const float*)d_in[8];
  const float* Wk = (const float*)d_in[9],  *bk = (const float*)d_in[10];
  const float* Wv = (const float*)d_in[11], *bv = (const float*)d_in[12];
  const float* Wg = (const float*)d_in[13], *bg = (const float*)d_in[14];
  const float* cw = (const float*)d_in[15], *cb = (const float*)d_in[16];
  const float* Wp = (const float*)d_in[17], *bp = (const float*)d_in[18];
  const float* Wpr = (const float*)d_in[19], *bpr = (const float*)d_in[20];
  const float* Wo = (const float*)d_in[21], *bo = (const float*)d_in[22];

  char* ws = (char*)d_ws;
  unsigned short* qb = (unsigned short*)(ws + OFF_Q);
  unsigned short* kb = (unsigned short*)(ws + OFF_K);
  unsigned short* vb = (unsigned short*)(ws + OFF_V);
  float* gateb = (float*)(ws + OFF_G);
  float* biasb = (float*)(ws + OFF_BIAS);
  float* orib  = (float*)(ws + OFF_ORI);
  float* convb2 = (float*)(ws + OFF_CONV);
  float* houtb = (float*)(ws + OFF_HO);

  float* out1 = (float*)d_out;
  float* out2 = out1 + (size_t)BB * LL * CC;   // 589824

  k_ln_proj<<<BB * LL, 192, 0, stream>>>(x, ln1w, ln1b, Wq, bq, Wk, bk, Wv, bv,
                                         Wg, bg, qb, kb, vb, gateb);
  k_pair_bias<<<BB * LL * LL / 8, 256, 0, stream>>>(pairX, ln2w, ln2b, Wp, bp, biasb);
  k_qk<<<BB * 3 * HH * 1024 / 4, 128, 0, stream>>>(qb, kb, biasb, orib);
  k_conv<<<BB * HH * 1024, 256, 0, stream>>>(orib, cw, cb, convb2);
  k_attn<<<BB * 3 * HH * 32 / 2, 64, 0, stream>>>(orib, convb2, vb, gateb, houtb);
  k_out2<<<BB * LL * LL / 8, 256, 0, stream>>>(orib, convb2, Wpr, bpr, out2);
  k_out1<<<BB * LL, 192, 0, stream>>>(houtb, Wo, bo, x, g, out1);
}
